// GCN_11218454577334
// MI455X (gfx1250) — matrix operand order repaired and checked by reading, not compiled, not run on hardware
//
#include <hip/hip_runtime.h>
#include <hip/hip_bf16.h>
#include <math.h>

// ---------------------------------------------------------------------------
// GENConv (3 layers) + pooling + dense head for MI455X (gfx1250, wave32).
// All GEMMs via v_wmma_f32_16x16x32_f16 (fp32 accumulate).
// Edge softmax aggregation: two fused passes (WMMA edge-embed + gather +
// atomic scatter), never materializing the [E,F] message tensor.
// All fragment loads branch-free; all offsets 32-bit; gathers batched.
// ---------------------------------------------------------------------------

static constexpr int   kN    = 50000;
static constexpr int   kE    = 1600000;
static constexpr int   kFin  = 128;
static constexpr int   kEdim = 16;
static constexpr int   kG    = 512;
static constexpr int   kOut  = 10;
static constexpr float kMsgEps = 1e-7f;
static constexpr float kSmEps  = 1e-16f;
static constexpr float kBnEps  = 1e-5f;

typedef __attribute__((ext_vector_type(16))) _Float16 v16h;
typedef __attribute__((ext_vector_type(8)))  float    v8f;

// ---------------- WMMA fragment helpers (wave32, ISA 7.12.2 layouts) -------

// A fragment 16x32 f16 from row-major [M,K] fp32. K % 32 == 0.
// Two 8-float runs per lane -> two global_load_b128 each.
__device__ __forceinline__ v16h frag_a_full(const float* __restrict__ A, int lda,
                                            int row0, int k0) {
  const int lane = threadIdx.x & 31;
  const unsigned off = (unsigned)(row0 + (lane & 15)) * (unsigned)lda +
                       (unsigned)(k0 + ((lane < 16) ? 0 : 16));
  const float* __restrict__ ar = A + off;
  v16h a;
#pragma unroll
  for (int j = 0; j < 8; ++j) {
    a[j]     = (_Float16)ar[j];
    a[j + 8] = (_Float16)ar[8 + j];
  }
  return a;
}

// B fragment 32x16 f16 from row-major [K,N] fp32, no bounds checks.
__device__ __forceinline__ v16h frag_b_full(const float* __restrict__ B, int ldb,
                                            int k0, int col0) {
  const int lane = threadIdx.x & 31;
  const unsigned off = (unsigned)(k0 + ((lane < 16) ? 0 : 16)) * (unsigned)ldb +
                       (unsigned)(col0 + (lane & 15));
  const float* __restrict__ bp = B + off;
  v16h b;
#pragma unroll
  for (int j = 0; j < 16; ++j) b[j] = (_Float16)bp[(unsigned)j * (unsigned)ldb];
  return b;
}

// B fragment with N-guard (only the Nc=10 head GEMM): clamp column so loads
// stay in bounds, then select zero -> v_cndmask, no EXEC manipulation.
__device__ __forceinline__ v16h frag_b_nguard(const float* __restrict__ B, int ldb,
                                              int k0, int col0, int Nvalid) {
  const int lane = threadIdx.x & 31;
  const int col  = col0 + (lane & 15);
  const int colc = (col < Nvalid) ? col : (Nvalid - 1);
  const bool ok  = (col < Nvalid);
  const unsigned off = (unsigned)(k0 + ((lane < 16) ? 0 : 16)) * (unsigned)ldb +
                       (unsigned)colc;
  const float* __restrict__ bp = B + off;
  v16h b;
#pragma unroll
  for (int j = 0; j < 16; ++j) {
    const float v = bp[(unsigned)j * (unsigned)ldb];
    b[j] = (_Float16)(ok ? v : 0.0f);
  }
  return b;
}

// ---------------- generic WMMA GEMM: C = A[M,K] @ B[K,Nc] + bias -----------
// M % 16 == 0, K % 32 == 0. NFULL additionally requires Nc % 16 == 0.

template <bool NFULL>
__global__ __launch_bounds__(128)
void wmma_gemm_bias(const float* __restrict__ A, const float* __restrict__ B,
                    const float* __restrict__ bias, float* __restrict__ C,
                    int M, int K, int Nc, int relu) {
  const int wpb = blockDim.x >> 5;
  const int wid = blockIdx.x * wpb + (threadIdx.x >> 5);
  const int ntm = M >> 4;
  const int ntn = (Nc + 15) >> 4;
  if (wid >= ntm * ntn) return;              // uniform per wave; EXEC stays full
  const int tm = wid % ntm;                  // tm fastest: waves in a block share
  const int tn = wid / ntm;                  // the same B column tile (L0 hits)
  const int row0 = tm << 4;
  const int col0 = tn << 4;

  v8f acc = {};
  for (int k0 = 0; k0 < K; k0 += 32) {
    if (k0 + 32 < K)                          // gfx1250 global_prefetch_b8 path
      __builtin_prefetch(A + (unsigned)row0 * (unsigned)K + (unsigned)(k0 + 32), 0, 1);
    const v16h a = frag_a_full(A, K, row0, k0);
    const v16h b = NFULL ? frag_b_full(B, Nc, k0, col0)
                         : frag_b_nguard(B, Nc, k0, col0, Nc);
    acc = __builtin_amdgcn_wmma_f32_16x16x32_f16(false, a, false, b,
                                                 (short)0, acc, false, false);
  }

  const int lane = threadIdx.x & 31;
  const int col  = col0 + (lane & 15);
  const int rb   = (lane < 16) ? 0 : 8;
  if (NFULL || col < Nc) {
    const float bv = bias ? bias[col] : 0.0f;
#pragma unroll
    for (int r = 0; r < 8; ++r) {
      float v = acc[r] + bv;
      if (relu) v = fmaxf(v, 0.0f);
      C[(unsigned)(row0 + rb + r) * (unsigned)Nc + (unsigned)col] = v;
    }
  }
}

// ------- fused edge kernels: WMMA edge-embed + gather + atomic scatter -----
// msg = relu(xs[src] + edge_attr @ We + be) + 1e-7   (msg > 0 always)
// K = 16, zero-padded to one 16x16x32 WMMA with a compile-time pattern:
// A elements j>=8 are zero; B elements j>=8 (K=16..31) select zero.

template <int F>
__device__ __forceinline__ v8f edge_embed_tile(const float* __restrict__ eattr,
                                               const float* __restrict__ We,
                                               int e0, int f0) {
  const int lane = threadIdx.x & 31;
  const bool lo  = (lane < 16);

  const float* __restrict__ ar =
      eattr + (unsigned)(e0 + (lane & 15)) * (unsigned)kEdim + (lo ? 0u : 8u);
  v16h a;
#pragma unroll
  for (int j = 0; j < 8; ++j) {
    a[j]     = (_Float16)ar[j];       // merges into global_load_b128 x2
    a[j + 8] = (_Float16)0.0f;        // K=16..31 zero pad
  }

  const float* __restrict__ bp = We + (unsigned)(f0 + (lane & 15));
  v16h b;
#pragma unroll
  for (int j = 0; j < 16; ++j) {
    const float v = bp[(unsigned)(((lo ? 0 : 8) + (j & 7)) * F)];      // rows 0..15 always in bounds
    b[j] = (_Float16)((j < 8) ? v : 0.0f);           // elements j>=8 are K=16..31 -> 0
  }

  v8f c = {};
  c = __builtin_amdgcn_wmma_f32_16x16x32_f16(false, a, false, b,
                                             (short)0, c, false, false);
  return c;
}

// Pass 1: per-destination running max (int atomicMax on positive-float bits).
template <int F>
__global__ __launch_bounds__(128)
void edge_softmax_max(const float* __restrict__ eattr, const float* __restrict__ We,
                      const float* __restrict__ be, const float* __restrict__ xs,
                      const int* __restrict__ src, const int* __restrict__ dst,
                      int* __restrict__ mmax_bits) {
  constexpr int ntf = F >> 4;
  const int wpb = blockDim.x >> 5;
  const int wid = blockIdx.x * wpb + (threadIdx.x >> 5);
  if (wid >= (kE >> 4) * ntf) return;
  const int ft = wid % ntf, et = wid / ntf;   // ft fastest: edge tile reuse in L0
  const int e0 = et << 4, f0 = ft << 4;

  const v8f c = edge_embed_tile<F>(eattr, We, e0, f0);

  const int lane = threadIdx.x & 31;
  const int feat = f0 + (lane & 15);
  const int rb   = (lane < 16) ? 0 : 8;
  const float bv = be[feat];

  int sA[8], dA[8];
#pragma unroll
  for (int r = 0; r < 8; ++r) {               // contiguous -> b128 x2 each
    sA[r] = src[e0 + rb + r];
    dA[r] = dst[e0 + rb + r];
  }
  float xv[8];
#pragma unroll
  for (int r = 0; r < 8; ++r)                 // 8 independent gathers, batched
    xv[r] = xs[(unsigned)sA[r] * (unsigned)F + (unsigned)feat];
#pragma unroll
  for (int r = 0; r < 8; ++r) {
    float m = fmaxf(c[r] + bv + xv[r], 0.0f) + kMsgEps;
    atomicMax(&mmax_bits[(unsigned)dA[r] * (unsigned)F + (unsigned)feat],
              __float_as_int(m));
  }
}

// Pass 2: recompute msg, accumulate exp-sum and weighted sum.
template <int F>
__global__ __launch_bounds__(128)
void edge_softmax_accum(const float* __restrict__ eattr, const float* __restrict__ We,
                        const float* __restrict__ be, const float* __restrict__ xs,
                        const int* __restrict__ src, const int* __restrict__ dst,
                        const float* __restrict__ mmax,
                        float* __restrict__ den, float* __restrict__ num) {
  constexpr int ntf = F >> 4;
  const int wpb = blockDim.x >> 5;
  const int wid = blockIdx.x * wpb + (threadIdx.x >> 5);
  if (wid >= (kE >> 4) * ntf) return;
  const int ft = wid % ntf, et = wid / ntf;
  const int e0 = et << 4, f0 = ft << 4;

  const v8f c = edge_embed_tile<F>(eattr, We, e0, f0);

  const int lane = threadIdx.x & 31;
  const int feat = f0 + (lane & 15);
  const int rb   = (lane < 16) ? 0 : 8;
  const float bv = be[feat];

  int sA[8], dA[8];
#pragma unroll
  for (int r = 0; r < 8; ++r) {
    sA[r] = src[e0 + rb + r];
    dA[r] = dst[e0 + rb + r];
  }
  float xv[8];
  unsigned di[8];
#pragma unroll
  for (int r = 0; r < 8; ++r) {
    xv[r] = xs[(unsigned)sA[r] * (unsigned)F + (unsigned)feat];
    di[r] = (unsigned)dA[r] * (unsigned)F + (unsigned)feat;
  }
  float mx[8];
#pragma unroll
  for (int r = 0; r < 8; ++r) mx[r] = mmax[di[r]];    // batched gathers
#pragma unroll
  for (int r = 0; r < 8; ++r) {
    const float m  = fmaxf(c[r] + bv + xv[r], 0.0f) + kMsgEps;
    const float ex = __expf(m - mx[r]);
    atomicAdd(&den[di[r]], ex);
    atomicAdd(&num[di[r]], ex * m);
  }
}

// ---------------- elementwise / reduction kernels --------------------------

__global__ void fill_f32(float* __restrict__ p, float v, size_t n) {
  const size_t i = (size_t)blockIdx.x * blockDim.x + threadIdx.x;
  if (i < n) p[i] = v;
}

__global__ void finalize_aggr(const float* __restrict__ num, const float* __restrict__ den,
                              const float* __restrict__ xd, float* __restrict__ h,
                              unsigned n) {
  const unsigned i = blockIdx.x * blockDim.x + threadIdx.x;
  if (i < n) h[i] = num[i] / (den[i] + kSmEps) + xd[i];
}

// One block per column: biased mean/var over N rows.
__global__ void bn_stats(const float* __restrict__ h, int N, int C,
                         float* __restrict__ mu, float* __restrict__ var) {
  const int c = blockIdx.x;
  float s = 0.0f, s2 = 0.0f;
  for (int n = threadIdx.x; n < N; n += blockDim.x) {
    const float v = h[(unsigned)n * (unsigned)C + (unsigned)c];
    s += v; s2 += v * v;
  }
  __shared__ float sh[256];
  __shared__ float sh2[256];
  sh[threadIdx.x] = s; sh2[threadIdx.x] = s2;
  __syncthreads();
  for (int o = blockDim.x >> 1; o > 0; o >>= 1) {
    if ((int)threadIdx.x < o) { sh[threadIdx.x] += sh[threadIdx.x + o];
                                sh2[threadIdx.x] += sh2[threadIdx.x + o]; }
    __syncthreads();
  }
  if (threadIdx.x == 0) {
    const float m = sh[0] / (float)N;
    mu[c] = m;
    var[c] = sh2[0] / (float)N - m * m;
  }
}

__global__ void bn_apply_relu(float* __restrict__ h, const float* __restrict__ g,
                              const float* __restrict__ bt, const float* __restrict__ mu,
                              const float* __restrict__ var, unsigned n, int C) {
  const unsigned i = blockIdx.x * blockDim.x + threadIdx.x;
  if (i >= n) return;
  const unsigned c = i % (unsigned)C;          // C is 128/256 -> strength-reduced
  const float v = (h[i] - mu[c]) * rsqrtf(var[c] + kBnEps) * g[c] + bt[c];
  h[i] = fmaxf(v, 0.0f);
}

__global__ void pool_accum(const float* __restrict__ h, const int* __restrict__ batch,
                           float* __restrict__ psum, float* __restrict__ pcnt) {
  const unsigned i = blockIdx.x * blockDim.x + threadIdx.x;
  if (i >= (unsigned)kN * 128u) return;
  const unsigned n = i >> 7;                   // C = 128 fixed
  const unsigned c = i & 127u;
  const int gi = batch[n];
  atomicAdd(&psum[(unsigned)gi * 128u + c], h[i]);
  if (c == 0) atomicAdd(&pcnt[gi], 1.0f);
}

__global__ void pool_divide(float* __restrict__ psum, const float* __restrict__ pcnt) {
  const unsigned i = blockIdx.x * blockDim.x + threadIdx.x;
  if (i >= (unsigned)(kG * 128)) return;
  psum[i] /= fmaxf(pcnt[i >> 7], 1.0f);
}

__global__ void log_softmax_rows(const float* __restrict__ logits, float* __restrict__ out) {
  const int g = blockIdx.x * blockDim.x + threadIdx.x;
  if (g >= kG) return;
  const float* row = logits + (unsigned)g * (unsigned)kOut;
  float mx = row[0];
#pragma unroll
  for (int j = 1; j < kOut; ++j) mx = fmaxf(mx, row[j]);
  float s = 0.0f;
#pragma unroll
  for (int j = 0; j < kOut; ++j) s += __expf(row[j] - mx);
  const float lse = mx + __logf(s);
#pragma unroll
  for (int j = 0; j < kOut; ++j) out[(unsigned)g * (unsigned)kOut + j] = row[j] - lse;
}

// ---------------- host orchestration ---------------------------------------

extern "C" void kernel_launch(void* const* d_in, const int* in_sizes, int n_in,
                              void* d_out, int out_size, void* d_ws, size_t ws_size,
                              hipStream_t stream) {
  (void)in_sizes; (void)n_in; (void)out_size; (void)ws_size;

  // --- input unpack (jax tree order; None params dropped for layer 2) ---
  const float* x      = (const float*)d_in[0];    // [N,128]
  const int*   eidx   = (const int*)  d_in[1];    // [2,E]
  const float* eattr  = (const float*)d_in[2];    // [E,16]
  const int*   batch  = (const int*)  d_in[3];    // [N]
  const int* src = eidx;
  const int* dst = eidx + kE;

  int p = 4;
  // layer 1: 128 -> 64 (full 12-tuple)
  const float* Ws1 = (const float*)d_in[p++]; const float* bs1 = (const float*)d_in[p++];
  const float* Wd1l = (const float*)d_in[p++]; const float* bd1l = (const float*)d_in[p++];
  const float* We1 = (const float*)d_in[p++]; const float* be1 = (const float*)d_in[p++];
  const float* W11 = (const float*)d_in[p++]; const float* b11 = (const float*)d_in[p++];
  const float* g1  = (const float*)d_in[p++]; const float* bt1 = (const float*)d_in[p++];
  const float* W21 = (const float*)d_in[p++]; const float* b21 = (const float*)d_in[p++];
  // layer 2: 64 -> 64 (no lin_src/lin_dst)
  const float* We2 = (const float*)d_in[p++]; const float* be2 = (const float*)d_in[p++];
  const float* W12 = (const float*)d_in[p++]; const float* b12 = (const float*)d_in[p++];
  const float* g2  = (const float*)d_in[p++]; const float* bt2 = (const float*)d_in[p++];
  const float* W22 = (const float*)d_in[p++]; const float* b22 = (const float*)d_in[p++];
  // layer 3: 64 -> 128 (full 12-tuple)
  const float* Ws3 = (const float*)d_in[p++]; const float* bs3 = (const float*)d_in[p++];
  const float* Wd3 = (const float*)d_in[p++]; const float* bd3 = (const float*)d_in[p++];
  const float* We3 = (const float*)d_in[p++]; const float* be3 = (const float*)d_in[p++];
  const float* W13 = (const float*)d_in[p++]; const float* b13 = (const float*)d_in[p++];
  const float* g3  = (const float*)d_in[p++]; const float* bt3 = (const float*)d_in[p++];
  const float* W23 = (const float*)d_in[p++]; const float* b23 = (const float*)d_in[p++];
  // dense head
  const float* Wh1 = (const float*)d_in[p++]; const float* bh1 = (const float*)d_in[p++];
  const float* Wh2 = (const float*)d_in[p++]; const float* bh2 = (const float*)d_in[p++];

  // --- workspace layout (floats); HMID [N,256] aliases XS∪XD -------------
  float* W = (float*)d_ws;
  const size_t NF = (size_t)kN * 128;
  float* XS   = W;                 // [N,128]
  float* XD   = W + NF;            // [N,128]
  float* HMID = XS;                // [N,256] alias (XS/XD dead by then)
  float* B0   = W + 2 * NF;        // seg-max bits, later aggr h
  float* B1   = W + 3 * NF;        // den
  float* B2   = W + 4 * NF;        // num
  float* HA   = W + 5 * NF;        // layer outputs (ping)
  float* HB   = W + 6 * NF;        // layer outputs (pong)
  float* SM   = W + 7 * NF;        // small region
  float* BNMU = SM;                // [256]
  float* BNVR = SM + 256;          // [256]
  float* PSUM = SM + 512;          // [512,128]
  float* PCNT = PSUM + (size_t)kG * 128; // [512]
  float* T1   = PCNT + kG;         // [512,64]
  float* LG   = T1 + (size_t)kG * 64;    // [512,10]

  auto gemm = [&](const float* A, const float* Bm, const float* bias, float* C,
                  int M, int K, int Nc, int relu) {
    const int ntn = (Nc + 15) / 16;
    const int waves = (M / 16) * ntn;
    const int blocks = (waves + 3) / 4;
    if ((Nc & 15) == 0)
      wmma_gemm_bias<true><<<blocks, 128, 0, stream>>>(A, Bm, bias, C, M, K, Nc, relu);
    else
      wmma_gemm_bias<false><<<blocks, 128, 0, stream>>>(A, Bm, bias, C, M, K, Nc, relu);
  };
  auto fill = [&](float* ptr, float v, size_t n) {
    fill_f32<<<(unsigned)((n + 255) / 256), 256, 0, stream>>>(ptr, v, n);
  };

  auto run_layer = [&](const float* xin, int Fi, int F,
                       const float* Wsrc, const float* bsrc,
                       const float* Wdst, const float* bdst,
                       const float* We, const float* be,
                       const float* W1, const float* b1,
                       const float* g, const float* bt,
                       const float* W2, const float* b2,
                       float* hout) {
    const float *xs, *xd;
    if (Wsrc) {
      gemm(xin, Wsrc, bsrc, XS, kN, Fi, F, 0);
      gemm(xin, Wdst, bdst, XD, kN, Fi, F, 0);
      xs = XS; xd = XD;
    } else {
      xs = xin; xd = xin;
    }
    // zero mmax/den/num (contiguous B0..B2)
    fill(B0, 0.0f, 3 * NF);
    const int ntf = F / 16;
    const int eblocks = ((kE / 16) * ntf + 3) / 4;
    if (F == 64) {
      edge_softmax_max<64><<<eblocks, 128, 0, stream>>>(eattr, We, be, xs, src, dst,
                                                        (int*)B0);
      edge_softmax_accum<64><<<eblocks, 128, 0, stream>>>(eattr, We, be, xs, src, dst,
                                                          B0, B1, B2);
    } else {
      edge_softmax_max<128><<<eblocks, 128, 0, stream>>>(eattr, We, be, xs, src, dst,
                                                         (int*)B0);
      edge_softmax_accum<128><<<eblocks, 128, 0, stream>>>(eattr, We, be, xs, src, dst,
                                                           B0, B1, B2);
    }
    const unsigned nf = (unsigned)kN * (unsigned)F;
    finalize_aggr<<<(nf + 255) / 256, 256, 0, stream>>>(B2, B1, xd, B0, nf);
    // MLP: Linear(F, 2F) -> BN -> ReLU -> Linear(2F, F); outer ReLU fused
    gemm(B0, W1, b1, HMID, kN, F, 2 * F, 0);
    bn_stats<<<2 * F, 256, 0, stream>>>(HMID, kN, 2 * F, BNMU, BNVR);
    const unsigned nh = (unsigned)kN * (unsigned)(2 * F);
    bn_apply_relu<<<(nh + 255) / 256, 256, 0, stream>>>(HMID, g, bt, BNMU, BNVR,
                                                        nh, 2 * F);
    gemm(HMID, W2, b2, hout, kN, 2 * F, F, /*relu=*/1);
  };

  run_layer(x,  kFin, 64,  Ws1, bs1, Wd1l, bd1l, We1, be1, W11, b11, g1, bt1, W21, b21, HA);
  run_layer(HA, 64,   64,  nullptr, nullptr, nullptr, nullptr,
                            We2, be2, W12, b12, g2, bt2, W22, b22, HB);
  run_layer(HB, 64,   128, Ws3, bs3, Wd3, bd3, We3, be3, W13, b13, g3, bt3, W23, b23, HA);

  // global mean pool over graphs
  fill(PSUM, 0.0f, (size_t)kG * 128 + kG);   // PSUM + PCNT contiguous
  pool_accum<<<((unsigned)kN * 128u + 255) / 256, 256, 0, stream>>>(HA, batch,
                                                                    PSUM, PCNT);
  pool_divide<<<(kG * 128 + 255) / 256, 256, 0, stream>>>(PSUM, PCNT);

  // dense head: [512,128]@[128,64] -> [512,64]@[64,10] -> log_softmax
  gemm(PSUM, Wh1, bh1, T1, kG, 128, 64, 0);
  gemm(T1,   Wh2, bh2, LG, kG, 64, kOut, 0);
  log_softmax_rows<<<(kG + 255) / 256, 256, 0, stream>>>(LG, (float*)d_out);
}
